// LossLinkerE2E_24790551232625
// MI455X (gfx1250) — matrix operand order repaired and checked by reading, not compiled, not run on hardware
//
#include <hip/hip_runtime.h>
#include <hip/hip_bf16.h>

// Problem constants (from reference setup_inputs)
#define BB 8
#define NN 4096
#define SS 512
#define CC 16
#define DD 1024
#define EE 256
#define HH 1024

typedef __attribute__((ext_vector_type(16))) __bf16 v16bf;
typedef __attribute__((ext_vector_type(8)))  float  v8f;

union FragBF {
    v16bf v;
    uint4 u[2];
};

// ---------------------------------------------------------------------------
// Kernel A: convert W1 (row-major (D+E) x H, f32) into transposed bf16 blocks:
//   W1sT[h][d] = W1[d][h]        (H x D)
//   W1eT[h][e] = W1[D+e][h]      (H x E)
// Transposed-bf16 layout makes each WMMA B-fragment two contiguous
// 16-byte per-lane loads.
// ---------------------------------------------------------------------------
__global__ __launch_bounds__(256) void convert_w1(const float* __restrict__ W1,
                                                  __bf16* __restrict__ W1sT,
                                                  __bf16* __restrict__ W1eT) {
    int tid = blockIdx.x * blockDim.x + threadIdx.x;
    const int totalS = HH * DD;
    const int totalE = HH * EE;
    if (tid < totalS) {
        int h = tid / DD, d = tid % DD;
        W1sT[tid] = (__bf16)W1[(size_t)d * HH + h];
    } else if (tid < totalS + totalE) {
        int t2 = tid - totalS;
        int h = t2 / EE, e = t2 % EE;
        W1eT[t2] = (__bf16)W1[(size_t)(DD + e) * HH + h];
    }
}

// ---------------------------------------------------------------------------
// Kernel B: Hspan[g][h] = span_vecs[b, linker_indices[g], :] @ W1s   (g = b*S+s)
// 256 blocks x 16 rows. 256 threads = 8 waves; each wave owns 128 H-columns
// (8 accumulator tiles). K loop over D in steps of 32 via wmma bf16.
// ---------------------------------------------------------------------------
__global__ __launch_bounds__(256) void span_gemm(const float* __restrict__ span_vecs,
                                                 const int*   __restrict__ linker_indices,
                                                 const __bf16* __restrict__ W1sT,
                                                 float* __restrict__ Hspan) {
    __shared__ __bf16 Abuf[16 * DD];   // 32 KB: 16 gathered rows, bf16
    __shared__ int lidx[16];

    const int t = threadIdx.x;
    const int rowbase = blockIdx.x * 16;

    if (t < 16) lidx[t] = linker_indices[rowbase + t];
    __syncthreads();

    // Stage A: gather 16 span rows (f32 -> bf16), coalesced.
    for (int i = 0; i < 64; ++i) {
        int flat = t + i * 256;
        int row = flat >> 10, col = flat & (DD - 1);
        int g = rowbase + row;
        int b = g >> 9;                          // S = 512
        const float* src = span_vecs + ((size_t)b * NN + lidx[row]) * DD;
        Abuf[flat] = (__bf16)src[col];
    }
    __syncthreads();

    const int lane = t & 31;
    const int wave = t >> 5;
    const int l15 = lane & 15;
    const int half = lane >> 4;                  // 0: lanes 0-15, 1: lanes 16-31
    const int colbase0 = wave * 128;

    v8f acc[8];
    for (int i = 0; i < 8; ++i)
        for (int j = 0; j < 8; ++j) acc[i][j] = 0.0f;

    for (int k0 = 0; k0 < DD; k0 += 32) {
        // A fragment (16x32 bf16): lanes 0-15 hold K {0..7,16..23}, lanes 16-31 {8..15,24..31}
        FragBF a;
        int abase = l15 * DD + k0 + (half << 3);
        a.u[0] = *(const uint4*)&Abuf[abase];
        a.u[1] = *(const uint4*)&Abuf[abase + 16];
        for (int tile = 0; tile < 8; ++tile) {
            int ncol = colbase0 + tile * 16 + l15;
            // B fragment (32x16 bf16) from transposed weights: lanes 0-15 K {0..7,16..23}, lanes 16-31 {8..15,24..31}
            const __bf16* bp = W1sT + (size_t)ncol * DD + k0 + (half << 3);
            FragBF bf;
            bf.u[0] = *(const uint4*)bp;
            bf.u[1] = *(const uint4*)(bp + 16);
            acc[tile] = __builtin_amdgcn_wmma_f32_16x16x32_bf16(
                false, a.v, false, bf.v, (short)0, acc[tile], false, false);
        }
    }

    // Store: C layout -> lanes 0-15: M=r, lanes 16-31: M=r+8, N = lane&15
    for (int tile = 0; tile < 8; ++tile) {
        int ncol = colbase0 + tile * 16 + l15;
        for (int r = 0; r < 8; ++r) {
            int g = rowbase + r + (half << 3);
            Hspan[(size_t)g * HH + ncol] = acc[tile][r];
        }
    }
}

// ---------------------------------------------------------------------------
// Kernel C: one block per linker position g = b*S+s. Gathers 16 candidate
// entity rows -> LDS bf16; 16x1024 GEMM over E=256 via wmma; fused epilogue:
// relu(h_cand + Hspan_row + b1) . W2 + b2, masked BCE loss, argmax.
// ---------------------------------------------------------------------------
__global__ __launch_bounds__(256) void cand_fused(const float* __restrict__ ent_table,
                                                  const __bf16* __restrict__ W1eT,
                                                  const float* __restrict__ Hspan,
                                                  const float* __restrict__ b1,
                                                  const float* __restrict__ W2,
                                                  const float* __restrict__ b2p,
                                                  const float* __restrict__ targets_all,
                                                  const int*   __restrict__ linker_indices,
                                                  const int*   __restrict__ candidates_all,
                                                  const int*   __restrict__ cand_len_all,
                                                  float* __restrict__ loss_out,
                                                  float* __restrict__ scores_out,
                                                  float* __restrict__ pred_out) {
    __shared__ __bf16 Abuf[16 * EE];   // 8 KB: 16 candidate rows, bf16
    __shared__ float  hb[HH];          // Hspan row + b1
    __shared__ float  w2s[HH];
    __shared__ float  wred[8 * 16];
    __shared__ float  fscore[16];
    __shared__ int    cands[16];
    __shared__ int    idx_sh;

    const int t = threadIdx.x;
    const int g = blockIdx.x;          // 0..B*S-1
    const int b = g >> 9;              // S = 512

    if (t == 0) idx_sh = linker_indices[g];
    __syncthreads();
    const int n = idx_sh;

    if (t < 16) cands[t] = candidates_all[((size_t)b * NN + n) * CC + t];
    for (int i = t; i < HH; i += 256) {
        hb[i]  = Hspan[(size_t)g * HH + i] + b1[i];
        w2s[i] = W2[i];
    }
    __syncthreads();

    // Stage A: gather 16 entity rows (f32 -> bf16), coalesced within rows.
    for (int i = 0; i < 16; ++i) {
        int flat = t + i * 256;
        int row = flat >> 8, col = flat & (EE - 1);
        Abuf[flat] = (__bf16)ent_table[(size_t)cands[row] * EE + col];
    }
    __syncthreads();

    const int lane = t & 31;
    const int wave = t >> 5;
    const int l15 = lane & 15;
    const int half = lane >> 4;
    const int colbase0 = wave * 128;

    v8f acc[8];
    for (int i = 0; i < 8; ++i)
        for (int j = 0; j < 8; ++j) acc[i][j] = 0.0f;

    for (int k0 = 0; k0 < EE; k0 += 32) {
        FragBF a;
        int abase = l15 * EE + k0 + (half << 3);
        a.u[0] = *(const uint4*)&Abuf[abase];
        a.u[1] = *(const uint4*)&Abuf[abase + 16];
        for (int tile = 0; tile < 8; ++tile) {
            int ncol = colbase0 + tile * 16 + l15;
            const __bf16* bp = W1eT + (size_t)ncol * EE + k0 + (half << 3);
            FragBF bf;
            bf.u[0] = *(const uint4*)bp;
            bf.u[1] = *(const uint4*)(bp + 16);
            acc[tile] = __builtin_amdgcn_wmma_f32_16x16x32_bf16(
                false, a.v, false, bf.v, (short)0, acc[tile], false, false);
        }
    }

    // Epilogue: relu(h_cand + hb) * W2, partial sums per candidate row.
    float partial[8];
    for (int r = 0; r < 8; ++r) partial[r] = 0.0f;
    for (int tile = 0; tile < 8; ++tile) {
        int ncol = colbase0 + tile * 16 + l15;
        float hbv = hb[ncol];
        float w2v = w2s[ncol];
        for (int r = 0; r < 8; ++r) {
            float v = acc[tile][r] + hbv;
            v = fmaxf(v, 0.0f);
            partial[r] += v * w2v;
        }
    }

    // Reduce across the 16 lanes sharing the same candidate row (width-16 butterfly).
    for (int r = 0; r < 8; ++r) {
        float v = partial[r];
        v += __shfl_xor(v, 8, 16);
        v += __shfl_xor(v, 4, 16);
        v += __shfl_xor(v, 2, 16);
        v += __shfl_xor(v, 1, 16);
        if (l15 == 0) wred[wave * 16 + r + (half << 3)] = v;
    }
    __syncthreads();

    if (t < 16) {
        float sc = b2p[0];
        for (int w = 0; w < 8; ++w) sc += wred[w * 16 + t];
        scores_out[(size_t)g * CC + t] = sc;
        fscore[t] = sc;
    }
    __syncthreads();

    if (t == 0) {
        int len = cand_len_all[(size_t)b * NN + n];
        const float* tgt = targets_all + ((size_t)b * NN + n) * CC;
        float lsum = 0.0f;
        int best = 0;
        float bestv = -3.0e38f;
        for (int c = 0; c < CC; ++c) {
            float sc = fscore[c];
            float m = (c < len) ? 1.0f : 0.0f;
            float bce = fmaxf(sc, 0.0f) - sc * tgt[c] + log1pf(expf(-fabsf(sc)));
            lsum += m * bce;
            float ms = sc - (1.0f - m) * 1e23f;
            if (ms > bestv) { bestv = ms; best = c; }
        }
        atomicAdd(loss_out, lsum);
        pred_out[g] = (float)best;
    }
}

// ---------------------------------------------------------------------------
// Launch
// ---------------------------------------------------------------------------
extern "C" void kernel_launch(void* const* d_in, const int* in_sizes, int n_in,
                              void* d_out, int out_size, void* d_ws, size_t ws_size,
                              hipStream_t stream) {
    const float* span_vecs = (const float*)d_in[0];
    const float* ent_table = (const float*)d_in[1];
    const float* W1        = (const float*)d_in[2];
    const float* b1        = (const float*)d_in[3];
    const float* W2        = (const float*)d_in[4];
    const float* b2        = (const float*)d_in[5];
    const float* targets   = (const float*)d_in[6];
    const int*   linker    = (const int*)d_in[7];
    const int*   cands     = (const int*)d_in[8];
    const int*   clens     = (const int*)d_in[9];

    float* out = (float*)d_out;
    // d_out layout: [0] loss, [1 .. B*S*C] scores, [1+B*S*C .. ] predictions (as float)
    float* loss_out   = out;
    float* scores_out = out + 1;
    float* pred_out   = out + 1 + BB * SS * CC;

    char* ws = (char*)d_ws;
    __bf16* W1sT = (__bf16*)ws;                                      // 2 MB
    __bf16* W1eT = (__bf16*)(ws + (size_t)HH * DD * 2);              // 0.5 MB
    float*  Hspan = (float*)(ws + (size_t)HH * DD * 2 + (size_t)HH * EE * 2); // 16 MB

    hipMemsetAsync(loss_out, 0, sizeof(float), stream);

    {
        int total = HH * DD + HH * EE;
        convert_w1<<<(total + 255) / 256, 256, 0, stream>>>(W1, W1sT, W1eT);
    }
    span_gemm<<<(BB * SS) / 16, 256, 0, stream>>>(span_vecs, linker, W1sT, Hspan);
    cand_fused<<<BB * SS, 256, 0, stream>>>(ent_table, W1eT, Hspan, b1, W2, b2,
                                            targets, linker, cands, clens,
                                            loss_out, scores_out, pred_out);
}
